// GraphPatchClassifier_52501680226658
// MI455X (gfx1250) — matrix operand order repaired and checked by reading, not compiled, not run on hardware
//
#include <hip/hip_runtime.h>
#include <hip/hip_bf16.h>
#include <math.h>
#include <stdint.h>

#define N_NODES 20000
#define IN_DIM  1536
#define HID     256
#define HEADS   4
#define HOPS    3
#define NCLS    3
#define NEDGE   320000
#define EPAD    (NEDGE + N_NODES)
#define LN_EPS  1e-5f
#define NEG_SLOPE 0.2f

typedef __bf16 bf16;
typedef __attribute__((ext_vector_type(16))) __bf16 v16bf;
typedef __attribute__((ext_vector_type(8)))  float  v8f;
typedef __attribute__((ext_vector_type(4)))  unsigned int u32x4;
typedef __attribute__((ext_vector_type(8)))  int i32x8;
typedef __attribute__((ext_vector_type(4)))  int i32x4;

#define BM 128
#define BN 64
#define BK 32
#define NC 256          // all GEMMs in this net have 256 output columns (compile-time!)
#define LDS_STRIDE 40   // bf16 elems per LDS row: 32 data + 8 pad = 80B (16B-aligned, bank-conflict-free)

#if defined(__has_builtin)
#if __has_builtin(__builtin_amdgcn_tensor_load_to_lds)
#define USE_TDM 1
#endif
#endif

struct Frag32B { uint4 a, b; };

#if defined(USE_TDM)
// ---------------------------------------------------------------------------
// TDM: async DMA of a [rows x kElems] bf16 tile (row stride = strideElems)
// into LDS at ldsOff, padding each 64B row to 80B (pad_interval=16dw, amount=4dw).
// Rows >= rowsValid read as zero (OOB returns zero per ISA Ch.8).
// Descriptor layout per CDNA5 ISA (6-arg builtin).
// ---------------------------------------------------------------------------
__device__ __forceinline__ void tdm_load_tile_bf16(const bf16* gsrc, unsigned ldsOff,
                                                   unsigned tileRows, unsigned rowsValid,
                                                   unsigned kElems, unsigned strideElems) {
  unsigned long long ga = (unsigned long long)(uintptr_t)gsrc;
  u32x4 g0;
  g0[0] = 1u;                                            // count=1, load, user mode
  g0[1] = ldsOff;                                        // lds_addr
  g0[2] = (unsigned)ga;                                  // global_addr[31:0]
  g0[3] = (unsigned)((ga >> 32) & 0x01FFFFFFu) | (2u << 30); // global_addr[56:32], type=2
  unsigned d0 = (1u << 16)      // data_size = 2 bytes
              | (1u << 20)      // pad_enable
              | (3u << 22)      // pad_interval: 16 dwords (64B) of data
              | (3u << 25);     // pad_amount: 4 dwords (16B) padding
  unsigned td0 = strideElems;   // tensor_dim0 (elements)
  unsigned td1 = rowsValid;     // tensor_dim1 (rows; OOB rows -> zero)
  unsigned d1 = (td0 & 0xFFFFu) << 16;
  unsigned d2 = ((td0 >> 16) & 0xFFFFu) | ((td1 & 0xFFFFu) << 16);
  unsigned d3 = ((td1 >> 16) & 0xFFFFu) | ((kElems & 0xFFFFu) << 16); // tile_dim0
  unsigned d4 = (tileRows & 0xFFFFu);                    // tile_dim1; tile_dim2=0
  unsigned d5 = strideElems;                             // tensor_dim0_stride[31:0]
  i32x8 g1;
  g1[0] = (int)d0; g1[1] = (int)d1; g1[2] = (int)d2; g1[3] = (int)d3;
  g1[4] = (int)d4; g1[5] = (int)d5; g1[6] = 0; g1[7] = 0;
  i32x4 z4; z4[0] = 0; z4[1] = 0; z4[2] = 0; z4[3] = 0;
  i32x8 z8;
  #pragma unroll
  for (int i = 0; i < 8; ++i) z8[i] = 0;
  __builtin_amdgcn_tensor_load_to_lds(g0, g1, z4, z4, z8, 0);
}

__device__ __forceinline__ void tdm_wait() {
#if __has_builtin(__builtin_amdgcn_s_wait_tensorcnt)
  __builtin_amdgcn_s_wait_tensorcnt(0);
#else
  asm volatile("s_wait_tensorcnt 0x0" ::: "memory");
#endif
}
#endif // USE_TDM

// ---------------------------------------------------------------------------
// WMMA GEMM: C[M,NC] = A[M,K](bf16,row) x B[K,NC](bf16,row) + bias (+GELU)
// 256 thr = 8 waves; wave (wm,wn) -> 32x32 tile via 2x2 v_wmma_f32_16x16x32_bf16.
// Double-buffered LDS: TDM(kt+1) + weight loads for kt+1 are issued BEFORE
// computing kt, so the async DMA overlaps the WMMAs. One barrier per k-tile.
// ---------------------------------------------------------------------------
__global__ __launch_bounds__(256)
void gemm_bias_act(const bf16* __restrict__ A, const bf16* __restrict__ B,
                   const float* __restrict__ bias, float* __restrict__ C,
                   int M, int K, int act) {
  __shared__ bf16 As[2][BM][LDS_STRIDE];   // row-major A tiles (padded rows)
  __shared__ bf16 Bst[2][BN][LDS_STRIDE];  // TRANSPOSED B tiles: Bst[buf][col][k]

  const int tid  = threadIdx.x;
  const int lane = tid & 31;
  const int wave = tid >> 5;
  const int wm   = wave & 3;
  const int wn   = wave >> 2;
  const int l16  = lane & 15;
  const int lh   = lane >> 4;

  const int rowBase = blockIdx.y * BM;
  const int colBase = blockIdx.x * BN;
  const int rowsValid = M - rowBase;

  v8f acc[2][2];
  #pragma unroll
  for (int i = 0; i < 2; ++i)
    #pragma unroll
    for (int j = 0; j < 2; ++j)
      #pragma unroll
      for (int r = 0; r < 8; ++r) acc[i][j][r] = 0.0f;

  const int bcol = tid & 63;           // B staging: col 0..63
  const int bk0  = (tid >> 6) * 8;     // 8 consecutive k each
  // Per-thread base pointer into B; advances by BK*NC per k-tile (const offsets!)
  const unsigned short* bptr =
      (const unsigned short*)B + (size_t)bk0 * NC + colBase + bcol;

#if defined(USE_TDM)
  const unsigned asOff[2] = { (unsigned)(size_t)(void*)&As[0][0][0],
                              (unsigned)(size_t)(void*)&As[1][0][0] };
#else
  const int ar = tid >> 1;             // A staging: row 0..127
  const int ac = (tid & 1) * 16;       // 16 bf16 (32B) each
#endif

  const int kTiles = K / BK;

  // ---- stage helpers -------------------------------------------------------
  auto stageB = [&](int kt, int buf) {
    const unsigned short* p = bptr + (size_t)kt * BK * NC;
    unsigned w[4];
    #pragma unroll
    for (int q = 0; q < 4; ++q) {
      unsigned lo = p[(2 * q) * NC];
      unsigned hi = p[(2 * q + 1) * NC];
      w[q] = lo | (hi << 16);
    }
    *(uint4*)&Bst[buf][bcol][bk0] = make_uint4(w[0], w[1], w[2], w[3]);
  };
#if !defined(USE_TDM)
  auto stageA = [&](int kt, int buf) {
    const int grow = rowBase + ar;
    if (grow < M) {
      const uint4* p4 = (const uint4*)(A + (size_t)grow * K + kt * BK + ac);
      *(uint4*)&As[buf][ar][ac]     = p4[0];
      *(uint4*)&As[buf][ar][ac + 8] = p4[1];
    } else {
      uint4 z = make_uint4(0u, 0u, 0u, 0u);
      *(uint4*)&As[buf][ar][ac]     = z;
      *(uint4*)&As[buf][ar][ac + 8] = z;
    }
  };
#endif

  // ---- prologue: tile 0 ----------------------------------------------------
#if defined(USE_TDM)
  if (wave == 0)
    tdm_load_tile_bf16(A + (size_t)rowBase * K, asOff[0], BM, (unsigned)rowsValid,
                       BK, (unsigned)K);
#else
  stageA(0, 0);
#endif
  stageB(0, 0);

  // ---- main loop: fetch kt+1 async while computing kt ----------------------
  for (int kt = 0; kt < kTiles; ++kt) {
    const int b  = kt & 1;
    const int b2 = b ^ 1;
#if defined(USE_TDM)
    if (wave == 0) tdm_wait();            // TDM(kt) complete
#endif
    __syncthreads();                      // B(kt) visible; compute(kt-1) done -> buf b2 free

    if (kt + 1 < kTiles) {
#if defined(USE_TDM)
      if (wave == 0)
        tdm_load_tile_bf16(A + (size_t)rowBase * K + (size_t)(kt + 1) * BK, asOff[b2],
                           BM, (unsigned)rowsValid, BK, (unsigned)K);
#else
      stageA(kt + 1, b2);
#endif
      stageB(kt + 1, b2);                 // overlaps compute(kt)
    }

    // Fragments as b128 pairs (CDNA5 16-bit WMMA layouts)
    v16bf afr[2], bfr[2];
    #pragma unroll
    for (int mt = 0; mt < 2; ++mt) {
      const int row = wm * 32 + mt * 16 + l16;
      Frag32B f;
      f.a = *(const uint4*)&As[b][row][lh * 8];        // K = lh*8 .. +7
      f.b = *(const uint4*)&As[b][row][16 + lh * 8];   // K = 16+lh*8 .. +7
      afr[mt] = __builtin_bit_cast(v16bf, f);
    }
    #pragma unroll
    for (int nt = 0; nt < 2; ++nt) {
      const int col = wn * 32 + nt * 16 + l16;
      Frag32B f;
      f.a = *(const uint4*)&Bst[b][col][lh * 8];       // K = lh*8 .. +7
      f.b = *(const uint4*)&Bst[b][col][16 + lh * 8];  // K = 16+lh*8 .. +7
      bfr[nt] = __builtin_bit_cast(v16bf, f);
    }

    #pragma unroll
    for (int mt = 0; mt < 2; ++mt)
      #pragma unroll
      for (int nt = 0; nt < 2; ++nt)
        acc[mt][nt] = __builtin_amdgcn_wmma_f32_16x16x32_bf16(
            false, afr[mt], false, bfr[nt], (short)0, acc[mt][nt], false, false);
  }

  // ---- store (C layout: VGPR r -> M = r + 8*laneHalf, N = lane%16) ---------
  #pragma unroll
  for (int mt = 0; mt < 2; ++mt) {
    #pragma unroll
    for (int nt = 0; nt < 2; ++nt) {
      const int col = colBase + wn * 32 + nt * 16 + l16;
      const float bv = bias ? bias[col] : 0.0f;
      #pragma unroll
      for (int r = 0; r < 8; ++r) {
        const int grow = rowBase + wm * 32 + mt * 16 + lh * 8 + r;
        if (grow < M) {
          float v = acc[mt][nt][r] + bv;
          if (act == 1) v = 0.5f * v * (1.0f + erff(v * 0.70710678118654752f));
          C[(size_t)grow * NC + col] = v;
        }
      }
    }
  }
}

// ---------------------------------------------------------------------------
// f32 -> bf16 bulk conversion
// ---------------------------------------------------------------------------
__global__ void f32_to_bf16_kernel(const float* __restrict__ in,
                                   unsigned short* __restrict__ out, long long n) {
  const long long stride = (long long)gridDim.x * blockDim.x;
  for (long long i = (long long)blockIdx.x * blockDim.x + threadIdx.x; i < n; i += stride) {
    bf16 b = (bf16)in[i];
    out[i] = __builtin_bit_cast(unsigned short, b);
  }
}

// ---------------------------------------------------------------------------
// LayerNorm (optional residual), emits f32 and bf16 copies.
// ---------------------------------------------------------------------------
__global__ __launch_bounds__(256)
void ln_kernel(const float* __restrict__ in, const float* __restrict__ res,
               const float* __restrict__ g, const float* __restrict__ b,
               float* __restrict__ out, unsigned short* __restrict__ out_bf, int M) {
  const int w    = (int)((blockIdx.x * blockDim.x + threadIdx.x) >> 5);
  const int lane = threadIdx.x & 31;
  if (w >= M) return;
  const float* pi = in + (size_t)w * HID;
  const float* pr = res ? res + (size_t)w * HID : nullptr;
  float vals[8];
  float s = 0.0f, s2 = 0.0f;
  #pragma unroll
  for (int j = 0; j < 8; ++j) {
    const int c = lane + 32 * j;
    float v = pi[c] + (pr ? pr[c] : 0.0f);
    vals[j] = v; s += v; s2 += v * v;
  }
  #pragma unroll
  for (int off = 16; off > 0; off >>= 1) {
    s  += __shfl_xor(s, off);
    s2 += __shfl_xor(s2, off);
  }
  const float mu  = s * (1.0f / HID);
  const float var = s2 * (1.0f / HID) - mu * mu;
  const float rs  = rsqrtf(var + LN_EPS);
  #pragma unroll
  for (int j = 0; j < 8; ++j) {
    const int c = lane + 32 * j;
    const float v = (vals[j] - mu) * rs * g[c] + b[c];
    out[(size_t)w * HID + c] = v;
    bf16 bb = (bf16)v;
    out_bf[(size_t)w * HID + c] = __builtin_bit_cast(unsigned short, bb);
  }
}

// ---------------------------------------------------------------------------
// Edge machinery
// ---------------------------------------------------------------------------
__global__ void build_edges_kernel(const long long* __restrict__ ei,
                                   int* __restrict__ src, int* __restrict__ dst) {
  const int idx = blockIdx.x * blockDim.x + threadIdx.x;
  if (idx >= EPAD) return;
  if (idx < NEDGE) {
    src[idx] = (int)ei[idx];
    dst[idx] = (int)ei[(size_t)NEDGE + idx];
  } else {
    src[idx] = idx - NEDGE;
    dst[idx] = idx - NEDGE;
  }
}

__global__ void gat_init_kernel(float* __restrict__ gout, const float* __restrict__ gbias,
                                float* __restrict__ m, float* __restrict__ denom) {
  const int idx = blockIdx.x * blockDim.x + threadIdx.x;
  if (idx < N_NODES * HID) gout[idx] = gbias[idx & (HID - 1)];
  if (idx < N_NODES * HEADS) { m[idx] = -1e30f; denom[idx] = 0.0f; }
}

__device__ __forceinline__ void atomicMaxFloat(float* addr, float val) {
  unsigned int* ua = (unsigned int*)addr;
  unsigned int old = __float_as_uint(*addr);
  while (__uint_as_float(old) < val) {
    unsigned int assumed = old;
    old = atomicCAS(ua, assumed, __float_as_uint(val));
    if (old == assumed) break;
  }
}

__global__ __launch_bounds__(256)
void edge_logits_kernel(const float* __restrict__ xl, const float* __restrict__ xr,
                        const int* __restrict__ src, const int* __restrict__ dst,
                        const float* __restrict__ att, float* __restrict__ logits,
                        float* __restrict__ m) {
  const int e    = (int)((blockIdx.x * blockDim.x + threadIdx.x) >> 5);
  const int lane = threadIdx.x & 31;
  if (e >= EPAD) return;
  const int s = src[e], d = dst[e];
  const float* pl = xl + (size_t)s * HID;
  const float* pr = xr + (size_t)d * HID;
  float hs[4] = {0.f, 0.f, 0.f, 0.f};
  #pragma unroll
  for (int j = 0; j < 8; ++j) {
    const int c = lane + 32 * j;       // head = j>>1
    float v = pl[c] + pr[c];
    v = v > 0.0f ? v : NEG_SLOPE * v;
    hs[j >> 1] += v * att[c];
  }
  #pragma unroll
  for (int off = 16; off > 0; off >>= 1)
    #pragma unroll
    for (int i = 0; i < 4; ++i) hs[i] += __shfl_xor(hs[i], off);
  if (lane == 0) {
    #pragma unroll
    for (int i = 0; i < 4; ++i) {
      logits[(size_t)e * HEADS + i] = hs[i];
      atomicMaxFloat(&m[d * HEADS + i], hs[i]);
    }
  }
}

__global__ void edge_exp_kernel(const float* __restrict__ logits, const int* __restrict__ dst,
                                const float* __restrict__ m, float* __restrict__ ex,
                                float* __restrict__ denom) {
  const int idx = blockIdx.x * blockDim.x + threadIdx.x;
  if (idx >= EPAD * HEADS) return;
  const int e = idx >> 2, hh = idx & 3;
  const int d = dst[e];
  const float v = expf(logits[idx] - m[d * HEADS + hh]);
  ex[idx] = v;
  atomicAdd(&denom[d * HEADS + hh], v);
}

__global__ void edge_scatter_kernel(const float* __restrict__ xl, const int* __restrict__ src,
                                    const int* __restrict__ dst, const float* __restrict__ ex,
                                    const float* __restrict__ denom, float* __restrict__ gout) {
  const long long total = (long long)EPAD * HID;
  for (long long idx = (long long)blockIdx.x * blockDim.x + threadIdx.x; idx < total;
       idx += (long long)gridDim.x * blockDim.x) {
    const int e  = (int)(idx >> 8);
    const int c  = (int)(idx & (HID - 1));
    const int hh = c >> 6;
    const int d = dst[e], s = src[e];
    const float alpha = ex[(size_t)e * HEADS + hh] / denom[d * HEADS + hh];
    atomicAdd(&gout[(size_t)d * HID + c], xl[(size_t)s * HID + c] * alpha);
  }
}

__global__ __launch_bounds__(256)
void head_kernel(const float* __restrict__ hid, const float* __restrict__ W2,
                 const float* __restrict__ b2, float* __restrict__ out) {
  const int w    = (int)((blockIdx.x * blockDim.x + threadIdx.x) >> 5);
  const int lane = threadIdx.x & 31;
  if (w >= N_NODES) return;
  const float* p = hid + (size_t)w * HID;
  float acc[NCLS] = {0.f, 0.f, 0.f};
  #pragma unroll
  for (int j = 0; j < 8; ++j) {
    const int c = lane + 32 * j;
    const float v = p[c];
    #pragma unroll
    for (int k = 0; k < NCLS; ++k) acc[k] += v * W2[c * NCLS + k];
  }
  #pragma unroll
  for (int off = 16; off > 0; off >>= 1)
    #pragma unroll
    for (int k = 0; k < NCLS; ++k) acc[k] += __shfl_xor(acc[k], off);
  if (lane == 0)
    #pragma unroll
    for (int k = 0; k < NCLS; ++k) out[(size_t)w * NCLS + k] = acc[k] + b2[k];
}

// ---------------------------------------------------------------------------
extern "C" void kernel_launch(void* const* d_in, const int* in_sizes, int n_in,
                              void* d_out, int out_size, void* d_ws, size_t ws_size,
                              hipStream_t stream) {
  const float*     features   = (const float*)d_in[0];
  const long long* edge_index = (const long long*)d_in[1];
  const float* proj_W = (const float*)d_in[2];
  const float* proj_b = (const float*)d_in[3];
  const float* n0_g   = (const float*)d_in[4];
  const float* n0_b   = (const float*)d_in[5];
  const float* Wl     = (const float*)d_in[6];
  const float* bl     = (const float*)d_in[7];
  const float* Wr     = (const float*)d_in[8];
  const float* br     = (const float*)d_in[9];
  const float* att    = (const float*)d_in[10];
  const float* gbias  = (const float*)d_in[11];
  const float* ln_g   = (const float*)d_in[12];
  const float* ln_b   = (const float*)d_in[13];
  const float* h1_W   = (const float*)d_in[14];
  const float* h1_b   = (const float*)d_in[15];
  const float* h2_W   = (const float*)d_in[16];
  const float* h2_b   = (const float*)d_in[17];
  float* out = (float*)d_out;

  char* ws = (char*)d_ws;
  auto carve = [&](size_t bytes) -> char* {
    char* p = ws; ws += (bytes + 255) & ~(size_t)255; return p;
  };
  float* x      = (float*)carve((size_t)N_NODES * HID * sizeof(float));
  float* xl     = (float*)carve((size_t)N_NODES * HID * sizeof(float));
  float* xr     = (float*)carve((size_t)N_NODES * HID * sizeof(float));
  float* gout   = (float*)carve((size_t)N_NODES * HID * sizeof(float));
  float* logits = (float*)carve((size_t)EPAD * HEADS * sizeof(float));
  float* ex     = (float*)carve((size_t)EPAD * HEADS * sizeof(float));
  float* mbuf   = (float*)carve((size_t)N_NODES * HEADS * sizeof(float));
  float* denom  = (float*)carve((size_t)N_NODES * HEADS * sizeof(float));
  int*   srcI   = (int*)carve((size_t)EPAD * sizeof(int));
  int*   dstI   = (int*)carve((size_t)EPAD * sizeof(int));
  unsigned short* feat_bf  = (unsigned short*)carve((size_t)N_NODES * IN_DIM * 2);
  unsigned short* x_bf     = (unsigned short*)carve((size_t)N_NODES * HID * 2);
  unsigned short* projW_bf = (unsigned short*)carve((size_t)IN_DIM * HID * 2);
  unsigned short* Wl_bf    = (unsigned short*)carve((size_t)HOPS * HID * HID * 2);
  unsigned short* Wr_bf    = (unsigned short*)carve((size_t)HOPS * HID * HID * 2);
  unsigned short* h1W_bf   = (unsigned short*)carve((size_t)HID * HID * 2);

  const dim3 gemmGrid(HID / BN, (N_NODES + BM - 1) / BM);
  const dim3 blk256(256);

  // One-time (per launch) conversions
  f32_to_bf16_kernel<<<2048, blk256, 0, stream>>>(features, feat_bf, (long long)N_NODES * IN_DIM);
  f32_to_bf16_kernel<<<512,  blk256, 0, stream>>>(proj_W, projW_bf, (long long)IN_DIM * HID);
  f32_to_bf16_kernel<<<512,  blk256, 0, stream>>>(Wl, Wl_bf, (long long)HOPS * HID * HID);
  f32_to_bf16_kernel<<<512,  blk256, 0, stream>>>(Wr, Wr_bf, (long long)HOPS * HID * HID);
  f32_to_bf16_kernel<<<256,  blk256, 0, stream>>>(h1_W, h1W_bf, (long long)HID * HID);
  build_edges_kernel<<<(EPAD + 255) / 256, blk256, 0, stream>>>(edge_index, srcI, dstI);

  // Input projection + LN
  gemm_bias_act<<<gemmGrid, blk256, 0, stream>>>((const bf16*)feat_bf, (const bf16*)projW_bf,
                                                 proj_b, gout, N_NODES, IN_DIM, 0);
  ln_kernel<<<(N_NODES * 32 + 255) / 256, blk256, 0, stream>>>(gout, nullptr, n0_g, n0_b,
                                                               x, x_bf, N_NODES);

  // GATv2 hops
  for (int h = 0; h < HOPS; ++h) {
    const bf16* Wl_h = (const bf16*)(Wl_bf + (size_t)h * HID * HID);
    const bf16* Wr_h = (const bf16*)(Wr_bf + (size_t)h * HID * HID);
    gemm_bias_act<<<gemmGrid, blk256, 0, stream>>>((const bf16*)x_bf, Wl_h, bl + h * HID, xl,
                                                   N_NODES, HID, 0);
    gemm_bias_act<<<gemmGrid, blk256, 0, stream>>>((const bf16*)x_bf, Wr_h, br + h * HID, xr,
                                                   N_NODES, HID, 0);
    gat_init_kernel<<<(N_NODES * HID + 255) / 256, blk256, 0, stream>>>(
        gout, gbias + h * HID, mbuf, denom);
    edge_logits_kernel<<<((size_t)EPAD * 32 + 255) / 256, blk256, 0, stream>>>(
        xl, xr, srcI, dstI, att + h * HID, logits, mbuf);
    edge_exp_kernel<<<(EPAD * HEADS + 255) / 256, blk256, 0, stream>>>(
        logits, dstI, mbuf, ex, denom);
    edge_scatter_kernel<<<4096, blk256, 0, stream>>>(xl, srcI, dstI, ex, denom, gout);
    ln_kernel<<<(N_NODES * 32 + 255) / 256, blk256, 0, stream>>>(
        gout, x, ln_g + h * HID, ln_b + h * HID, x, x_bf, N_NODES);
  }

  // Head: GELU MLP + tiny classifier
  gemm_bias_act<<<gemmGrid, blk256, 0, stream>>>((const bf16*)x_bf, (const bf16*)h1W_bf,
                                                 h1_b, gout, N_NODES, HID, 1);
  head_kernel<<<(N_NODES * 32 + 255) / 256, blk256, 0, stream>>>(gout, h2_W, h2_b, out);
}
